// MetaHeteroLinear_49847390437447
// MI455X (gfx1250) — matrix operand order repaired and checked by reading, not compiled, not run on hardware
//
#include <hip/hip_runtime.h>

typedef __attribute__((ext_vector_type(16))) _Float16 v16h;
typedef __attribute__((ext_vector_type(8)))  float    v8f;
typedef __attribute__((ext_vector_type(2)))  float    v2f;

#define IN_C   128
#define OUT_C  128
#define MEMSZ  512
#define HID    256
#define NTYPE  8

// ---- workspace layout (bytes) ----
// meta (u32): [0..7]=cnt  [8..15]=cursor  [16..23]=segBase  [24..32]=tilePrefix(9)
#define OFF_BIAS   192u       // 8*128 f32      (4096 B)
#define OFF_H1W    4352u      // 8*256 f32      (8192 B)
#define OFF_H1B    12544u
#define OFF_H2W    20736u
#define OFF_H2B    28928u
#define OFF_WHI    37120u     // 8*16384 f16, transposed [t][n][k]  (256 KB)
#define OFF_WLO    299264u    // f16 residual                        (256 KB)
#define OFF_PERM   561408u    // N u32

__global__ void k_init(unsigned* meta) {
    if (threadIdx.x < 8) meta[threadIdx.x] = 0u;
}

// One hidden layer for BOTH generator MLPs: block b -> (type=b>>1, which=b&1)
__global__ void k_hid(const float* __restrict__ inw, const float* __restrict__ inb,
                      const float* __restrict__ ww,  const float* __restrict__ wb,
                      const float* __restrict__ bw,  const float* __restrict__ bb,
                      float* __restrict__ ow, float* __restrict__ ob, int K) {
    int b = blockIdx.x;
    int t = b >> 1, which = b & 1;
    int j = threadIdx.x;                       // 0..255 hidden unit
    const float* in = (which ? inb : inw) + t * K;
    const float* w  = which ? wb : ww;
    float s = (which ? bb : bw)[j];
    for (int k = 0; k < K; ++k) s = fmaf(in[k], w[k * HID + j], s);
    (which ? ob : ow)[t * HID + j] = fmaxf(s, 0.f);
}

// Final layer of weight generator: fp32 result split into f16 hi+lo, stored
// transposed [t][n][k] so WMMA B-fragments are contiguous per lane.
__global__ void k_wgen(const float* __restrict__ h2, const float* __restrict__ w3,
                       const float* __restrict__ b3,
                       _Float16* __restrict__ whi, _Float16* __restrict__ wlo) {
    int b = blockIdx.x;                        // 512 blocks: t = b>>6, o-chunk = b&63
    int t = b >> 6;
    int o = ((b & 63) << 8) + threadIdx.x;     // 0..16383 = k*128+n (reshape order)
    const float* h = h2 + t * HID;
    float s = b3[o];
    for (int kk = 0; kk < HID; ++kk) s = fmaf(h[kk], w3[kk * (IN_C * OUT_C) + o], s);
    int k = o >> 7, n = o & 127;
    _Float16 hi = (_Float16)s;
    _Float16 lo = (_Float16)(s - (float)hi);
    size_t d = ((size_t)t << 14) + ((size_t)n << 7) + (size_t)k;
    whi[d] = hi; wlo[d] = lo;
}

__global__ void k_bgen(const float* __restrict__ h2, const float* __restrict__ w3,
                       const float* __restrict__ b3, float* __restrict__ bias) {
    int idx = blockIdx.x * 256 + threadIdx.x;  // 1024 = 8*128
    int t = idx >> 7, n = idx & 127;
    const float* h = h2 + t * HID;
    float s = b3[n];
    for (int kk = 0; kk < HID; ++kk) s = fmaf(h[kk], w3[kk * OUT_C + n], s);
    bias[idx] = s;
}

// hierarchical histogram: LDS counts, 8 global atomics per block
__global__ void k_hist(const int* __restrict__ tv, int n, unsigned* __restrict__ meta) {
    __shared__ unsigned sh[8];
    int tid = threadIdx.x;
    if (tid < 8) sh[tid] = 0u;
    __syncthreads();
    int i = blockIdx.x * 256 + tid;
    if (i < n) atomicAdd(&sh[tv[i] & 7], 1u);
    __syncthreads();
    if (tid < 8 && sh[tid]) atomicAdd(&meta[tid], sh[tid]);
}

__global__ void k_prefix(unsigned* meta) {
    if (threadIdx.x != 0 || blockIdx.x != 0) return;
    unsigned base = 0, tiles = 0;
    for (int t = 0; t < 8; ++t) {
        meta[16 + t] = base;          // segBase
        meta[8 + t]  = base;          // cursor
        meta[24 + t] = tiles;         // tilePrefix (exclusive)
        base  += meta[t];
        tiles += (meta[t] + 15u) >> 4;
    }
    meta[32] = tiles;                 // total padded tiles
}

__global__ void k_scatter(const int* __restrict__ tv, int n,
                          unsigned* __restrict__ meta, unsigned* __restrict__ perm) {
    __shared__ unsigned sh[8];
    __shared__ unsigned sbase[8];
    int tid = threadIdx.x;
    if (tid < 8) sh[tid] = 0u;
    __syncthreads();
    int i = blockIdx.x * 256 + tid;
    unsigned r = 0; int t = 0;
    if (i < n) { t = tv[i] & 7; r = atomicAdd(&sh[t], 1u); }
    __syncthreads();
    if (tid < 8) sbase[tid] = sh[tid] ? atomicAdd(&meta[8 + tid], sh[tid]) : 0u;
    __syncthreads();
    if (i < n) perm[sbase[t] + r] = (unsigned)i;
}

// Main GEMM: 1 wave = one 16-node type-uniform tile x full 128 outputs.
// f16 hi/lo split GEMM: C += Ahi*Bhi + Alo*Bhi + Ahi*Blo  (fp32-class accuracy)
__global__ __launch_bounds__(128) void k_gemm(
        const float* __restrict__ x, const unsigned* __restrict__ meta,
        const _Float16* __restrict__ whi, const _Float16* __restrict__ wlo,
        const float* __restrict__ bias, const unsigned* __restrict__ perm,
        float* __restrict__ out) {
    unsigned total = meta[32];
    unsigned tile = blockIdx.x * 4u + (threadIdx.x >> 5);
    if (tile >= total) return;                       // wave-uniform exit

    unsigned t = 0;
    #pragma unroll
    for (unsigned i = 1; i < 8; ++i) if (tile >= meta[24 + i]) t = i;
    unsigned lt = tile - meta[24 + t];               // tile index within type
    unsigned c  = meta[t];                           // nodes of this type
    unsigned sb = meta[16 + t];                      // segment base in perm

    unsigned lane = threadIdx.x & 31u;
    unsigned m    = lane & 15u;                      // A row this lane owns
    bool     hi16 = lane >= 16u;
    unsigned rowIdx = lt * 16u + m;
    unsigned pidx   = rowIdx < c ? rowIdx : (c - 1u);
    unsigned nid    = perm[sb + pidx];               // lane i (i<16) holds row i's node
    bool     validm = rowIdx < c;
    const float* xrow = x + (size_t)nid * IN_C;

    // ---- A fragments for 4 K-steps, contiguous k = 16h + s layout (16x32 per step) ----
    v16h ahi[4], alo[4];
    #pragma unroll
    for (int kb = 0; kb < 4; ++kb) {
        #pragma unroll
        for (int v = 0; v < 8; ++v) {
            int kloc = (hi16 ? 16 : 0) + (v << 1);
            int k = kb * 32 + kloc;
            v2f p = *(const v2f*)(xrow + k);          // contiguous pair, 8B aligned
            if (!validm) { p.x = 0.f; p.y = 0.f; }
            _Float16 h0 = (_Float16)p.x;
            _Float16 h1 = (_Float16)p.y;
            ahi[kb][2 * v]     = h0;
            ahi[kb][2 * v + 1] = h1;
            alo[kb][2 * v]     = (_Float16)(p.x - (float)h0);
            alo[kb][2 * v + 1] = (_Float16)(p.y - (float)h1);
        }
    }

    const _Float16* wbh = whi + ((size_t)t << 14);
    const _Float16* wbl = wlo + ((size_t)t << 14);
    unsigned ncol0 = lane & 15u;
    unsigned kgrp  = hi16 ? 16u : 0u;

    #pragma unroll
    for (int nb = 0; nb < 8; ++nb) {
        v8f acc = {0.f, 0.f, 0.f, 0.f, 0.f, 0.f, 0.f, 0.f};
        unsigned n = nb * 16u + ncol0;
        size_t boff = ((size_t)n << 7) + kgrp;        // [t][n][k], 32B aligned
        #pragma unroll
        for (int kb = 0; kb < 4; ++kb) {
            v16h bhi = *(const v16h*)(wbh + boff + kb * 32);
            v16h blo = *(const v16h*)(wbl + boff + kb * 32);
            acc = __builtin_amdgcn_wmma_f32_16x16x32_f16(false, ahi[kb], false, bhi,
                                                         (short)0, acc, false, false);
            acc = __builtin_amdgcn_wmma_f32_16x16x32_f16(false, alo[kb], false, bhi,
                                                         (short)0, acc, false, false);
            acc = __builtin_amdgcn_wmma_f32_16x16x32_f16(false, ahi[kb], false, blo,
                                                         (short)0, acc, false, false);
        }
        float bv = bias[t * OUT_C + n];
        #pragma unroll
        for (int r = 0; r < 8; ++r) {                 // C layout: VGPR r -> row r(+8)
            unsigned mr = (unsigned)r + (hi16 ? 8u : 0u);
            unsigned nrow = (unsigned)__builtin_amdgcn_ds_bpermute((int)(mr << 2), (int)nid);
            if (lt * 16u + mr < c)
                out[(size_t)nrow * OUT_C + n] = acc[r] + bv;  // 64B contiguous/half-wave
        }
    }
}

extern "C" void kernel_launch(void* const* d_in, const int* in_sizes, int n_in,
                              void* d_out, int out_size, void* d_ws, size_t ws_size,
                              hipStream_t stream) {
    const float* x      = (const float*)d_in[0];
    const int*   tv     = (const int*)  d_in[1];
    const float* ef     = (const float*)d_in[2];
    const float* wg_w1  = (const float*)d_in[3];
    const float* wg_b1  = (const float*)d_in[4];
    const float* wg_w2  = (const float*)d_in[5];
    const float* wg_b2  = (const float*)d_in[6];
    const float* wg_w3  = (const float*)d_in[7];
    const float* wg_b3  = (const float*)d_in[8];
    const float* bg_w1  = (const float*)d_in[9];
    const float* bg_b1  = (const float*)d_in[10];
    const float* bg_w2  = (const float*)d_in[11];
    const float* bg_b2  = (const float*)d_in[12];
    const float* bg_w3  = (const float*)d_in[13];
    const float* bg_b3  = (const float*)d_in[14];

    int N = in_sizes[0] / IN_C;
    char* ws = (char*)d_ws;
    unsigned*  meta = (unsigned*)ws;
    float*     bias = (float*)(ws + OFF_BIAS);
    float*     h1w  = (float*)(ws + OFF_H1W);
    float*     h1b  = (float*)(ws + OFF_H1B);
    float*     h2w  = (float*)(ws + OFF_H2W);
    float*     h2b  = (float*)(ws + OFF_H2B);
    _Float16*  whi  = (_Float16*)(ws + OFF_WHI);
    _Float16*  wlo  = (_Float16*)(ws + OFF_WLO);
    unsigned*  perm = (unsigned*)(ws + OFF_PERM);

    k_init   <<<1,   32,  0, stream>>>(meta);
    k_hid    <<<16,  256, 0, stream>>>(ef,  ef,  wg_w1, bg_w1, wg_b1, bg_b1, h1w, h1b, MEMSZ);
    k_hid    <<<16,  256, 0, stream>>>(h1w, h1b, wg_w2, bg_w2, wg_b2, bg_b2, h2w, h2b, HID);
    k_wgen   <<<512, 256, 0, stream>>>(h2w, wg_w3, wg_b3, whi, wlo);
    k_bgen   <<<4,   256, 0, stream>>>(h2b, bg_w3, bg_b3, bias);

    int hb = (N + 255) / 256;
    k_hist   <<<hb, 256, 0, stream>>>(tv, N, meta);
    k_prefix <<<1,  1,   0, stream>>>(meta);
    k_scatter<<<hb, 256, 0, stream>>>(tv, N, meta, perm);

    int maxTiles = (N + 15) / 16 + NTYPE;            // upper bound on padded tiles
    k_gemm   <<<(maxTiles + 3) / 4, 128, 0, stream>>>(x, meta, whi, wlo, bias, perm,
                                                      (float*)d_out);
}
